// BCMGOOLSTM_60103772340471
// MI455X (gfx1250) — matrix operand order repaired and checked by reading, not compiled, not run on hardware
//
#include <hip/hip_runtime.h>
#include <hip/hip_bf16.h>

// Problem constants (from reference): T=1500, B=16, D=512, L=P=512, G=2048.
#define T_ 1500
#define B_ 16
#define D_ 512
#define L_ 512
#define P_ 512
#define G_ 2048
#define NWG 16          // persistent workgroups in the sequential kernel
#define LDSROW 520      // padded LDS row stride (bf16 elems): 1040B -> rotates banks

typedef __attribute__((ext_vector_type(16))) __bf16 v16bf;
typedef __attribute__((ext_vector_type(8)))  __bf16 v8bf;
typedef __attribute__((ext_vector_type(8)))  float  v8f;

// ---- workspace byte offsets (all 256B aligned) ----
#define OFF_WIH   ((size_t)0)           // bf16 W_ih   (G x D)          2,097,152 B
#define OFF_WHH   ((size_t)2097152)     // bf16 W_hh   (G x P)          2,097,152 B
#define OFF_WYM   ((size_t)4194304)     // bf16 wym    (P x L)            524,288 B
#define OFF_WYMT  ((size_t)4718592)     // bf16 wym^T  (L x P)            524,288 B
#define OFF_WFUSE ((size_t)5242880)     // bf16 W_f = W_hh*wym (G x L)  2,097,152 B
#define OFF_XBF   ((size_t)7340032)     // bf16 X      (T*B x D)       24,576,000 B
#define OFF_WOUT  ((size_t)31916032)    // bf16 w_out  (T*B x G)       98,304,000 B
#define OFF_HS    ((size_t)130220032)   // bf16 h[0..T](T+1 x B x L)   24,592,384 B
#define OFF_SYNC  ((size_t)154812416)   // u32 grid-barrier counter

__device__ __forceinline__ float sigmoidf_(float x) { return 1.0f / (1.0f + __expf(-x)); }

// One 16x16 output tile, K-loop of v_wmma_f32_16x16x32_bf16.
// A: 16 rows (lda stride), B: 16 "N rows" = weight rows, contiguous over K.
// A lane layout per ISA 7.12.2: lanes 0-15 hold K {0..7,16..23}, lanes 16-31 {8..15,24..31}.
__device__ __forceinline__ v8f mma_row_tile(const __bf16* __restrict__ A, int lda,
                                            const __bf16* __restrict__ B, int ldb,
                                            int K, v8f acc) {
  const int lane = threadIdx.x & 31;
  const int row  = lane & 15;
  const int asel = (lane >> 4) * 8;     // A half-select (interleaved 8-elem chunks)

  const __bf16* ap = A + (size_t)row * lda;
  const __bf16* bp = B + (size_t)row * ldb;
  for (int k0 = 0; k0 < K; k0 += 32) {
    union { v16bf v; v8bf h[2]; } a;
    a.h[0] = *(const v8bf*)(ap + k0 + asel);
    a.h[1] = *(const v8bf*)(ap + k0 + 16 + asel);
    union { v16bf v; v8bf h[2]; } bb;
    bb.h[0] = *(const v8bf*)(bp + k0 + asel);
    bb.h[1] = *(const v8bf*)(bp + k0 + 16 + asel);
    v16bf b = bb.v;
    acc = __builtin_amdgcn_wmma_f32_16x16x32_bf16(false, a.v, false, b,
                                                  (short)0, acc, false, false);
  }
  return acc;
}

// ---------------- gather / convert ----------------
__global__ void k_gather_w(const float* __restrict__ vec, const int* __restrict__ idx,
                           __bf16* __restrict__ W, int rows, int cols) {
  int t = blockIdx.x * blockDim.x + threadIdx.x;
  if (t >= rows * cols) return;
  int g = t / cols, d = t - g * cols;
  // idx laid out (cols, rows): W[g][d] = vec[idx[d][g]]
  W[(size_t)g * cols + d] = (__bf16)vec[idx[(size_t)d * rows + g]];
}

__global__ void k_prep_wym(const float* __restrict__ w, __bf16* __restrict__ wb,
                           __bf16* __restrict__ wt) {
  int t = blockIdx.x * blockDim.x + threadIdx.x;  // over P*L
  int p = t >> 9, l = t & 511;
  float v = w[t];
  wb[(size_t)p * L_ + l] = (__bf16)v;
  wt[(size_t)l * P_ + p] = (__bf16)v;
}

__global__ void k_f32_to_bf16(const float* __restrict__ in, __bf16* __restrict__ out, int n) {
  int t = blockIdx.x * blockDim.x + threadIdx.x;
  if (t < n) out[t] = (__bf16)in[t];
}

// ---------------- parallel GEMMs ----------------
// w_out = X * W_ih^T  (24000x2048, K=512), stored bf16.  grid (T, 16) x 256thr
__global__ void k_gemm_ih(const __bf16* __restrict__ X, const __bf16* __restrict__ Wih,
                          __bf16* __restrict__ Wout) {
  const int t  = blockIdx.x;
  const int nt = blockIdx.y * 8 + (threadIdx.x >> 5);  // 0..127
  v8f acc = {};
  acc = mma_row_tile(X + (size_t)t * (B_ * D_), D_,
                     Wih + (size_t)nt * 16 * D_, D_, D_, acc);
  const int lane = threadIdx.x & 31;
  const int n = lane & 15, mo = (lane >> 4) << 3;
  __bf16* C = Wout + (size_t)t * (B_ * G_) + (size_t)nt * 16 + n;
#pragma unroll
  for (int v = 0; v < 8; ++v) C[(size_t)(v + mo) * G_] = (__bf16)acc[v];
}

// W_f = W_hh * wym  (2048x512, K=512).  grid (128, 4) x 256thr
__global__ void k_gemm_fuse(const __bf16* __restrict__ Whh, const __bf16* __restrict__ WymT,
                            __bf16* __restrict__ Wf) {
  const int mt = blockIdx.x;
  const int nt = blockIdx.y * 8 + (threadIdx.x >> 5);  // 0..31
  v8f acc = {};
  acc = mma_row_tile(Whh + (size_t)mt * 16 * P_, P_,
                     WymT + (size_t)nt * 16 * P_, P_, P_, acc);
  const int lane = threadIdx.x & 31;
  const int n = lane & 15, mo = (lane >> 4) << 3;
  __bf16* C = Wf + (size_t)mt * 16 * L_ + (size_t)nt * 16 + n;
#pragma unroll
  for (int v = 0; v < 8; ++v) C[(size_t)(v + mo) * L_] = (__bf16)acc[v];
}

// ys[t] = h[t+1] * wym^T  (f32 out).  grid (T, 4) x 256thr
__global__ void k_gemm_out(const __bf16* __restrict__ Hs, const __bf16* __restrict__ Wym,
                           float* __restrict__ Y) {
  const int t  = blockIdx.x;
  const int nt = blockIdx.y * 8 + (threadIdx.x >> 5);  // 0..31
  v8f acc = {};
  acc = mma_row_tile(Hs + (size_t)(t + 1) * (B_ * L_), L_,
                     Wym + (size_t)nt * 16 * L_, L_, L_, acc);
  const int lane = threadIdx.x & 31;
  const int n = lane & 15, mo = (lane >> 4) << 3;
  float* C = Y + (size_t)t * (B_ * P_) + (size_t)nt * 16 + n;
#pragma unroll
  for (int v = 0; v < 8; ++v) C[(size_t)(v + mo) * P_] = acc[v];
}

// ---------------- sequential LSTM core ----------------
// 16 persistent WGs; WG j owns gate columns g = q*512 + j*32 + c (q=0..3, c=0..31),
// i.e. 128 fused-weight rows pinned in LDS (130KB of the 320KB WGP LDS).
// Per step: u-slice = h_prev * W_f^T (8 waves x 1 tile x 16 wmma), gates, write
// h slice, then an agent-scope atomic grid barrier to publish h for all WGs.
__global__ void __launch_bounds__(256, 1)
k_seq(const __bf16* __restrict__ Wf, const __bf16* __restrict__ Wout,
      const float* __restrict__ bias, __bf16* __restrict__ Hs,
      unsigned* __restrict__ syncp) {
  extern __shared__ char smem[];
  __bf16* lds_w = (__bf16*)smem;                                   // 128 x LDSROW
  float*  lds_u = (float*)(smem + (size_t)128 * LDSROW * 2);       // 4*16*32 f32
  float*  lds_c = (float*)(smem + (size_t)128 * LDSROW * 2 + 2048 * 4);  // c state 16x32

  const int j    = blockIdx.x;
  const int tid  = threadIdx.x;
  const int wave = tid >> 5;
  const int lane = tid & 31;

  // Stage this WG's 128 W_f rows into LDS (16B chunks), init cell state.
  for (int i = tid; i < 128 * (L_ / 8); i += 256) {
    int lr = i >> 6, ch = i & 63;
    int q = lr >> 5, c = lr & 31;
    const __bf16* src = Wf + (size_t)(q * 512 + j * 32 + c) * L_ + ch * 8;
    *(v8bf*)(lds_w + (size_t)lr * LDSROW + ch * 8) = *(const v8bf*)src;
  }
  for (int i = tid; i < 512; i += 256) lds_c[i] = 0.0f;
  __syncthreads();

  const int q = wave >> 1, r = wave & 1;                 // this wave's 16x16 u tile
  const __bf16* brow = lds_w + (size_t)(q * 32 + r * 16) * LDSROW;
  unsigned bar = 0;

  for (int t = 0; t < T_; ++t) {
    const __bf16* hprev = Hs + (size_t)t * (B_ * L_);
    v8f acc = {};
    acc = mma_row_tile(hprev, L_, brow, LDSROW, L_, acc);
    {
      const int n = lane & 15, mo = (lane >> 4) << 3;
#pragma unroll
      for (int v = 0; v < 8; ++v)
        lds_u[q * 512 + (v + mo) * 32 + r * 16 + n] = acc[v];
    }
    __syncthreads();

    __bf16* hnext = Hs + (size_t)(t + 1) * (B_ * L_);
    for (int e = tid; e < 512; e += 256) {
      int b = e >> 5, c = e & 31;
      int gcol = j * 32 + c;
      size_t wb = (size_t)t * (B_ * G_) + (size_t)b * G_ + gcol;
      float pf = (float)Wout[wb + 0 * 512] + bias[gcol + 0 * 512] + lds_u[0 * 512 + b * 32 + c];
      float pi = (float)Wout[wb + 1 * 512] + bias[gcol + 1 * 512] + lds_u[1 * 512 + b * 32 + c];
      float po = (float)Wout[wb + 2 * 512] + bias[gcol + 2 * 512] + lds_u[2 * 512 + b * 32 + c];
      float pc = (float)Wout[wb + 3 * 512] + bias[gcol + 3 * 512] + lds_u[3 * 512 + b * 32 + c];
      float ct = sigmoidf_(pi) * tanhf(pc) + sigmoidf_(pf) * lds_c[e];
      lds_c[e] = ct;
      hnext[(size_t)b * L_ + gcol] = (__bf16)(sigmoidf_(po) * tanhf(ct));
    }

    // device-wide barrier: publish h slice, wait for all 16 WGs
    __syncthreads();
    __threadfence();
    bar += NWG;
    if (tid == 0) {
      __hip_atomic_fetch_add(syncp, 1u, __ATOMIC_RELEASE, __HIP_MEMORY_SCOPE_AGENT);
      while (__hip_atomic_load(syncp, __ATOMIC_ACQUIRE, __HIP_MEMORY_SCOPE_AGENT) < bar)
        __builtin_amdgcn_s_sleep(2);
    }
    __syncthreads();
  }
}

extern "C" void kernel_launch(void* const* d_in, const int* in_sizes, int n_in,
                              void* d_out, int out_size, void* d_ws, size_t ws_size,
                              hipStream_t stream) {
  (void)in_sizes; (void)n_in; (void)out_size; (void)ws_size;
  const float* x    = (const float*)d_in[0];
  const float* vih  = (const float*)d_in[1];
  const float* vhh  = (const float*)d_in[2];
  const float* bias = (const float*)d_in[3];
  const float* wym  = (const float*)d_in[4];
  const int*   iih  = (const int*)d_in[5];
  const int*   ihh  = (const int*)d_in[6];

  char* ws = (char*)d_ws;
  __bf16* Wih  = (__bf16*)(ws + OFF_WIH);
  __bf16* Whh  = (__bf16*)(ws + OFF_WHH);
  __bf16* Wymb = (__bf16*)(ws + OFF_WYM);
  __bf16* WymT = (__bf16*)(ws + OFF_WYMT);
  __bf16* Wf   = (__bf16*)(ws + OFF_WFUSE);
  __bf16* Xbf  = (__bf16*)(ws + OFF_XBF);
  __bf16* Wout = (__bf16*)(ws + OFF_WOUT);
  __bf16* Hs   = (__bf16*)(ws + OFF_HS);
  unsigned* sy = (unsigned*)(ws + OFF_SYNC);

  // deterministic per-launch init (graph-capture safe)
  hipMemsetAsync(Hs, 0, (size_t)B_ * L_ * 2, stream);   // h0 = 0
  hipMemsetAsync(sy, 0, 256, stream);                   // barrier counter

  k_gather_w  <<<(G_ * D_) / 256, 256, 0, stream>>>(vih, iih, Wih, G_, D_);
  k_gather_w  <<<(G_ * P_) / 256, 256, 0, stream>>>(vhh, ihh, Whh, G_, P_);
  k_prep_wym  <<<(P_ * L_) / 256, 256, 0, stream>>>(wym, Wymb, WymT);
  k_f32_to_bf16<<<(T_ * B_ * D_) / 256, 256, 0, stream>>>(x, Xbf, T_ * B_ * D_);

  k_gemm_fuse <<<dim3(G_ / 16, 4), 256, 0, stream>>>(Whh, WymT, Wf);
  k_gemm_ih   <<<dim3(T_, G_ / 128), 256, 0, stream>>>(Xbf, Wih, Wout);

  size_t smem = (size_t)128 * LDSROW * 2 + 2048 * 4 + 512 * 4;  // ~143 KB of 320 KB
  k_seq       <<<NWG, 256, smem, stream>>>(Wf, Wout, bias, Hs, sy);

  k_gemm_out  <<<dim3(T_, 4), 256, 0, stream>>>(Hs, Wymb, (float*)d_out);
}
